// CrossBlock_65137474011240
// MI455X (gfx1250) — matrix operand order repaired and checked by reading, not compiled, not run on hardware
//
#include <hip/hip_runtime.h>
#include <math.h>

// ---------------------------------------------------------------------------
// CrossBlock for MI455X (gfx1250, wave32, WMMA + async global->LDS staging).
// All GEMMs use v_wmma_f32_16x16x32_bf16 (fp32 accumulate). Softmax in fp32.
// d_out layout (fp32): out0[2*2048*256] | out1[2*2048*256] |
//                      attn01[2*4*2048*2048] | attn10[2*4*2048*2048]
// Raw sim is staged in the attn01 region and finalized in place.
// ---------------------------------------------------------------------------

#define Bv 2
#define Nv 2048
#define Dv 256
#define Hv 4
#define DHv 64
#define ALPHA 0.01f
#define QK_SCALE 0.35355339059327373f   /* 64^-0.25 */

typedef __bf16 v16bf __attribute__((ext_vector_type(16)));
typedef float  v8f   __attribute__((ext_vector_type(8)));

__device__ __forceinline__ __bf16 f2bf(float f) {
  union { float f; unsigned u; } v; v.f = f;
  unsigned r = v.u + 0x7FFFu + ((v.u >> 16) & 1u);  // round-to-nearest-even
  union { unsigned short s; __bf16 b; } o; o.s = (unsigned short)(r >> 16);
  return o.b;
}

__device__ __forceinline__ v8f wmma_bf16(v16bf a, v16bf b, v8f c) {
  return __builtin_amdgcn_wmma_f32_16x16x32_bf16(false, a, false, b, (short)0, c,
                                                 false, false);
}

// Async copy of 16 bytes global -> LDS (GLOBAL_LOAD_ASYNC_TO_LDS_B128, ASYNCcnt).
// LDS aperture occupies the high 32 bits of the flat address, so the low 32
// bits of a __shared__ pointer are the LDS byte offset (ISA 10.2).
__device__ __forceinline__ void async_g2l_b128(void* lds_ptr, const void* gptr) {
  unsigned loff = (unsigned)(size_t)lds_ptr;
  asm volatile("global_load_async_to_lds_b128 %0, %1, off"
               :: "v"(loff), "v"(gptr) : "memory");
}
__device__ __forceinline__ void wait_async0() {
  asm volatile("s_wait_asynccnt 0x0" ::: "memory");
}

// A fragment: 16x32 bf16 tile, row-major source, leading dim lda.
// lane<16: M=lane, K = [k0..k0+7] and [k0+16..k0+23]
// lane>=16: M=lane-16, K = [k0+8..k0+15] and [k0+24..k0+31]
__device__ __forceinline__ v16bf load_a_bf16(const __bf16* __restrict__ A, int lda,
                                             int row0, int k0, int lane) {
  int m  = row0 + (lane & 15);
  int kh = k0 + ((lane >> 4) << 3);
  const __bf16* p = A + (size_t)m * lda + kh;
  v16bf a;
  ((float4*)&a)[0] = *(const float4*)p;         // 8 bf16
  ((float4*)&a)[1] = *(const float4*)(p + 16);  // 8 bf16
  return a;
}

// Same fragment but converting from an fp32 source (attention weights).
__device__ __forceinline__ v16bf load_a_f32(const float* __restrict__ A, int lda,
                                            int row0, int k0, int lane) {
  int m  = row0 + (lane & 15);
  int kh = k0 + ((lane >> 4) << 3);
  const float4* p = (const float4*)(A + (size_t)m * lda + kh);
  const float4* q = (const float4*)(A + (size_t)m * lda + kh + 16);
  float4 f0 = p[0], f1 = p[1], g0 = q[0], g1 = q[1];
  v16bf a;
  a[0]=f2bf(f0.x);  a[1]=f2bf(f0.y);  a[2]=f2bf(f0.z);  a[3]=f2bf(f0.w);
  a[4]=f2bf(f1.x);  a[5]=f2bf(f1.y);  a[6]=f2bf(f1.z);  a[7]=f2bf(f1.w);
  a[8]=f2bf(g0.x);  a[9]=f2bf(g0.y);  a[10]=f2bf(g0.z); a[11]=f2bf(g0.w);
  a[12]=f2bf(g1.x); a[13]=f2bf(g1.y); a[14]=f2bf(g1.z); a[15]=f2bf(g1.w);
  return a;
}

// B fragment: 32x16 (KxN) where B[k][n] = S[n][k] (S row-major, ld lds_).
// Used for QK^T (B = qk1 rows): two 16-byte loads per lane, 32 bytes apart.
__device__ __forceinline__ v16bf load_bT_bf16(const __bf16* __restrict__ S, int lds_,
                                              int col0, int k0, int lane) {
  int n  = col0 + (lane & 15);
  int kb = k0 + ((lane >> 4) << 3);
  const __bf16* p = S + (size_t)n * lds_ + kb;
  v16bf b;
  ((float4*)&b)[0] = *(const float4*)p;
  ((float4*)&b)[1] = *(const float4*)(p + 16);
  return b;
}

// B fragment from an LDS slab laid out [k][0..15] (16 bf16 per row).
__device__ __forceinline__ v16bf load_b_lds(const __bf16* Bl, int k0, int lane) {
  int n  = lane & 15;
  int kb = k0 + ((lane >> 4) << 3);
  v16bf b;
#pragma unroll
  for (int k = 0; k < 16; ++k) b[k] = Bl[(kb + k + ((k >> 3) << 3)) * 16 + n];
  return b;
}

// ---------------------------------------------------------------------------
// fp32 -> bf16 staging with arbitrary destination leading dim (for concat).
// ---------------------------------------------------------------------------
__global__ void cvt2bf(const float* __restrict__ src, __bf16* __restrict__ dst,
                       int cols, int ld, int total) {
  int i = blockIdx.x * blockDim.x + threadIdx.x;
  if (i >= total) return;
  int r = i / cols, c = i - r * cols;
  dst[(size_t)r * ld + c] = f2bf(src[i]);
}

// ---------------------------------------------------------------------------
// Generic WMMA GEMM: out = (A[M,K](bf16) @ Bw[K,Nc](bf16) + bias) * scale (+resid)
// mode 0: bf16 out   mode 1: f32 out   mode 2: bf16 head-split [B,H,N,DH]
// Block = 128 rows x 16 cols (8 waves); the shared K x 16 B-slab is staged into
// LDS once via GLOBAL_LOAD_ASYNC_TO_LDS_B128. Requires K <= 512, M % 128 == 0.
// ---------------------------------------------------------------------------
__global__ void gemm_wmma(const __bf16* __restrict__ A, int lda,
                          const __bf16* __restrict__ Bw, int ldb,
                          const float* __restrict__ bias,
                          const float* __restrict__ resid, int ldr,
                          void* __restrict__ out, int ldo, int col_off,
                          int M, int Nc, int K, float scale, int mode) {
  __shared__ __align__(16) __bf16 Bl[512 * 16];  // 16 KB
  int tiles_n = Nc >> 4;
  int rb = blockIdx.x / tiles_n;
  int tn = blockIdx.x - rb * tiles_n;
  int col0 = tn << 4;

  // stage B slab: rows k -> Bl[k*16 .. k*16+15]; 2 x 16B chunks per row
  for (int hr = threadIdx.x; hr < K * 2; hr += 256) {
    int k = hr >> 1, half = hr & 1;
    async_g2l_b128(Bl + k * 16 + half * 8,
                   Bw + (size_t)k * ldb + col0 + half * 8);
  }
  wait_async0();
  __syncthreads();

  int lane = threadIdx.x & 31;
  int row0 = rb * 128 + ((threadIdx.x >> 5) << 4);

  v8f c = {};
  for (int k = 0; k < K; k += 32) {
    v16bf a = load_a_bf16(A, lda, row0, k, lane);
    v16bf b = load_b_lds(Bl, k, lane);
    c = wmma_bf16(a, b, c);
  }

  int ccol  = col0 + (lane & 15);
  int rbase = row0 + ((lane >> 4) << 3);
  float bvv = bias ? bias[ccol] : 0.f;
#pragma unroll
  for (int r = 0; r < 8; ++r) {
    int m = rbase + r;
    float val = (c[r] + bvv) * scale;
    if (resid) val += resid[(size_t)m * ldr + ccol];
    if (mode == 0) {
      ((__bf16*)out)[(size_t)m * ldo + col_off + ccol] = f2bf(val);
    } else if (mode == 1) {
      ((float*)out)[(size_t)m * ldo + col_off + ccol] = val;
    } else {  // head-split: rows = b*2048+n, cols = h*64+d
      int b_ = m >> 11, n_ = m & 2047, h_ = ccol >> 6, d_ = ccol & 63;
      ((__bf16*)out)[(((size_t)(b_ * Hv + h_) * Nv + n_) << 6) + d_] = f2bf(val);
    }
  }
}

// ---------------------------------------------------------------------------
// sim[b,h] = qk0[b,h] (2048x64) @ qk1[b,h]^T  -> raw scores into attn01 region
// Both operands are direct global loads (two 16-byte pieces per lane); no staging needed.
// ---------------------------------------------------------------------------
__global__ void qk_sim_wmma(const __bf16* __restrict__ qk0,
                            const __bf16* __restrict__ qk1,
                            float* __restrict__ sim) {
  const int TP = 128 * 128;
  int wave = blockIdx.x * 8 + (threadIdx.x >> 5);
  int lane = threadIdx.x & 31;
  if (wave >= 8 * TP) return;
  int bh = wave / TP, rem = wave - bh * TP;
  int ti = rem >> 7, tj = rem & 127;
  const __bf16* Q  = qk0 + (size_t)bh * Nv * DHv;
  const __bf16* Kp = qk1 + (size_t)bh * Nv * DHv;
  float* S = sim + (size_t)bh * Nv * Nv;

  v8f c = {};
#pragma unroll
  for (int k = 0; k < DHv; k += 32) {
    v16bf a = load_a_bf16(Q, DHv, ti * 16, k, lane);
    v16bf b = load_bT_bf16(Kp, DHv, tj * 16, k, lane);
    c = wmma_bf16(a, b, c);
  }
  int col   = tj * 16 + (lane & 15);
  int rbase = ti * 16 + ((lane >> 4) << 3);
#pragma unroll
  for (int r = 0; r < 8; ++r) S[(size_t)(rbase + r) * Nv + col] = c[r];
}

// ---------------------------------------------------------------------------
// Softmax statistics. Online (max, sum) combine.
// ---------------------------------------------------------------------------
__device__ __forceinline__ void sm_acc(float& m, float& s, float x) {
  float M = fmaxf(m, x);
  s = s * expf(m - M) + expf(x - M);
  m = M;
}

__global__ void row_softmax_stats(const float* __restrict__ sim,
                                  const float* __restrict__ mask1,
                                  float* __restrict__ rmax, float* __restrict__ rinv) {
  int row = blockIdx.x;           // 0 .. B*H*N-1
  int b   = row >> 13;            // row / (H*N)
  const float* p  = sim + (size_t)row * Nv;
  const float* mk = mask1 + b * Nv;
  float m = -INFINITY, s = 0.f;
  for (int j = threadIdx.x; j < Nv; j += 256)
    if (mk[j] >= ALPHA) sm_acc(m, s, p[j]);
  __shared__ float sm_[256], ss_[256];
  sm_[threadIdx.x] = m; ss_[threadIdx.x] = s; __syncthreads();
  for (int off = 128; off > 0; off >>= 1) {
    if (threadIdx.x < off) {
      float m2 = sm_[threadIdx.x + off], s2 = ss_[threadIdx.x + off];
      float M = fmaxf(sm_[threadIdx.x], m2);
      float sn = (M == -INFINITY) ? 0.f
               : ss_[threadIdx.x] * expf(sm_[threadIdx.x] - M) + s2 * expf(m2 - M);
      sm_[threadIdx.x] = M; ss_[threadIdx.x] = sn;
    }
    __syncthreads();
  }
  if (threadIdx.x == 0) {
    rmax[row] = sm_[0];
    rinv[row] = ss_[0] > 0.f ? 1.f / ss_[0] : 0.f;
  }
}

__global__ void col_softmax_stats(const float* __restrict__ sim,
                                  const float* __restrict__ mask0,
                                  float* __restrict__ cmax, float* __restrict__ cinv) {
  // one block = (bh, 64-column slab); 4 threads per column, coalesced row reads
  int blk = blockIdx.x;
  int bh  = blk >> 5;
  int j0  = (blk & 31) << 6;
  int b   = bh >> 2;
  int tx = threadIdx.x & 63, ty = threadIdx.x >> 6;
  const float* base = sim + (size_t)bh * Nv * Nv;
  const float* mk   = mask0 + b * Nv;
  float m = -INFINITY, s = 0.f;
  for (int i = ty; i < Nv; i += 4)
    if (mk[i] >= ALPHA) sm_acc(m, s, base[(size_t)i * Nv + j0 + tx]);
  __shared__ float sm_[256], ss_[256];
  sm_[threadIdx.x] = m; ss_[threadIdx.x] = s; __syncthreads();
  if (ty == 0) {
#pragma unroll
    for (int t = 1; t < 4; ++t) {
      float m2 = sm_[tx + t * 64], s2 = ss_[tx + t * 64];
      float M = fmaxf(m, m2);
      if (M == -INFINITY) { s = 0.f; }
      else s = s * expf(m - M) + s2 * expf(m2 - M);
      m = M;
    }
    cmax[bh * Nv + j0 + tx] = m;
    cinv[bh * Nv + j0 + tx] = s > 0.f ? 1.f / s : 0.f;
  }
}

// ---------------------------------------------------------------------------
// Finalize: read raw sim once; write attn01 in place (coalesced) and attn10
// transposed through an LDS 32x32 tile (coalesced stores).
// ---------------------------------------------------------------------------
__global__ void softmax_finalize(float* __restrict__ attn01, float* __restrict__ attn10,
                                 const float* __restrict__ mask0,
                                 const float* __restrict__ mask1,
                                 const float* __restrict__ rmax, const float* __restrict__ rinv,
                                 const float* __restrict__ cmax, const float* __restrict__ cinv) {
  int t  = blockIdx.x;
  int bh = t >> 12, rem = t & 4095;
  int i0 = (rem >> 6) << 5, j0 = (rem & 63) << 5;
  int b  = bh >> 2;
  float* A01 = attn01 + (size_t)bh * Nv * Nv;
  float* A10 = attn10 + (size_t)bh * Nv * Nv;
  __shared__ float tile[32][33];
  int tx = threadIdx.x & 31, ty = threadIdx.x >> 5;
#pragma unroll
  for (int yy = 0; yy < 4; ++yy) {
    int i = i0 + ty + yy * 8, j = j0 + tx;
    float sv = A01[(size_t)i * Nv + j];
    bool bad1 = mask1[b * Nv + j] < ALPHA;
    bool bad0 = mask0[b * Nv + i] < ALPHA;
    float a01 = bad1 ? 0.f : expf(sv - rmax[bh * Nv + i]) * rinv[bh * Nv + i];
    float a10 = bad0 ? 0.f : expf(sv - cmax[bh * Nv + j]) * cinv[bh * Nv + j];
    A01[(size_t)i * Nv + j] = a01;
    tile[tx][ty + yy * 8] = a10;  // tile[local_j][local_i]
  }
  __syncthreads();
#pragma unroll
  for (int yy = 0; yy < 4; ++yy) {
    int jj = ty + yy * 8;
    A10[(size_t)(j0 + jj) * Nv + i0 + tx] = tile[jj][tx];
  }
}

// ---------------------------------------------------------------------------
// m = attn (f32 -> bf16 in-flight) @ V ; merged-head bf16 output [4096, 256].
// Block = (bh, 16-col tile, 128-row group): 8 waves. The V column slab is
// staged through LDS in two 1024-row chunks with async global->LDS copies.
// ---------------------------------------------------------------------------
__global__ void attn_v_wmma(const float* __restrict__ attn,
                            const __bf16* __restrict__ v,
                            __bf16* __restrict__ mm) {
  __shared__ __align__(16) __bf16 Bl[1024 * 16];  // 32 KB
  int blk = blockIdx.x;               // 8 * 4 * 16 = 512 blocks
  int bh = blk >> 6, rem = blk & 63;
  int tn = rem >> 4, rowgrp = rem & 15;
  const float*  A = attn + (size_t)bh * Nv * Nv;
  const __bf16* V = v + (size_t)bh * Nv * DHv;
  int col0 = tn << 4;
  int lane = threadIdx.x & 31;
  int row0 = rowgrp * 128 + ((threadIdx.x >> 5) << 4);

  v8f c = {};
  for (int kc = 0; kc < Nv; kc += 1024) {
    for (int hr = threadIdx.x; hr < 2048; hr += 256) {
      int k = hr >> 1, half = hr & 1;
      async_g2l_b128(Bl + k * 16 + half * 8,
                     V + (size_t)(kc + k) * DHv + col0 + half * 8);
    }
    wait_async0();
    __syncthreads();
    for (int k = 0; k < 1024; k += 32) {
      v16bf a = load_a_f32(A, Nv, row0, kc + k, lane);
      v16bf b = load_b_lds(Bl, k, lane);
      c = wmma_bf16(a, b, c);
    }
    __syncthreads();  // before re-staging the slab
  }

  int b_ = bh >> 2, h_ = bh & 3;
  int colg  = h_ * DHv + col0 + (lane & 15);
  int rbase = b_ * Nv + row0 + ((lane >> 4) << 3);
#pragma unroll
  for (int r = 0; r < 8; ++r)
    mm[(size_t)(rbase + r) * Dv + colg] = f2bf(c[r]);
}

// ---------------------------------------------------------------------------
// LayerNorm(512) + exact GELU, fp32 in -> bf16 out.
// ---------------------------------------------------------------------------
__global__ void ln_gelu(const float* __restrict__ h1, const float* __restrict__ g,
                        const float* __restrict__ bb, __bf16* __restrict__ h2) {
  int row = blockIdx.x;
  const float* p = h1 + (size_t)row * 512;
  float x0 = p[threadIdx.x], x1 = p[threadIdx.x + 256];
  __shared__ float red[256];
  red[threadIdx.x] = x0 + x1; __syncthreads();
  for (int off = 128; off > 0; off >>= 1) {
    if (threadIdx.x < off) red[threadIdx.x] += red[threadIdx.x + off];
    __syncthreads();
  }
  float mu = red[0] * (1.f / 512.f);
  __syncthreads();
  float d0 = x0 - mu, d1 = x1 - mu;
  red[threadIdx.x] = d0 * d0 + d1 * d1; __syncthreads();
  for (int off = 128; off > 0; off >>= 1) {
    if (threadIdx.x < off) red[threadIdx.x] += red[threadIdx.x + off];
    __syncthreads();
  }
  float rs = rsqrtf(red[0] * (1.f / 512.f) + 1e-5f);
#pragma unroll
  for (int t = 0; t < 2; ++t) {
    int c = threadIdx.x + t * 256;
    float x = (t == 0) ? x0 : x1;
    float y = (x - mu) * rs * g[c] + bb[c];
    float ge = 0.5f * y * (1.f + erff(y * 0.70710678118654752f));
    h2[(size_t)row * 512 + c] = f2bf(ge);
  }
}

// ---------------------------------------------------------------------------
extern "C" void kernel_launch(void* const* d_in, const int* in_sizes, int n_in,
                              void* d_out, int out_size, void* d_ws, size_t ws_size,
                              hipStream_t stream) {
  const float* x0  = (const float*)d_in[0];
  const float* x1  = (const float*)d_in[1];
  const float* cm0 = (const float*)d_in[2];
  const float* cm1 = (const float*)d_in[3];
  const float* Wqk = (const float*)d_in[4];
  const float* bqk = (const float*)d_in[5];
  const float* Wv  = (const float*)d_in[6];
  const float* bv  = (const float*)d_in[7];
  const float* Wo  = (const float*)d_in[8];
  const float* bo  = (const float*)d_in[9];
  const float* W1  = (const float*)d_in[10];
  const float* b1  = (const float*)d_in[11];
  const float* lng = (const float*)d_in[12];
  const float* lnb = (const float*)d_in[13];
  const float* W2  = (const float*)d_in[14];
  const float* b2  = (const float*)d_in[15];

  float* out  = (float*)d_out;
  float* out0    = out;                                   // [2,2048,256]
  float* out1    = out + (size_t)Bv * Nv * Dv;            // [2,2048,256]
  float* attn01  = out + (size_t)2 * Bv * Nv * Dv;        // [2,4,2048,2048]
  float* attn10  = attn01 + (size_t)Bv * Hv * Nv * Nv;    // [2,4,2048,2048]

  // ---- workspace carve-up ----------------------------------------------
  char* ws = (char*)d_ws;
  size_t off = 0;
  auto take = [&](size_t bytes) { char* p = ws + off; off += (bytes + 255) & ~(size_t)255; return p; };
  __bf16* wqkb = (__bf16*)take(65536 * 2);
  __bf16* wvb  = (__bf16*)take(65536 * 2);
  __bf16* wob  = (__bf16*)take(65536 * 2);
  __bf16* w1b  = (__bf16*)take(262144 * 2);
  __bf16* w2b  = (__bf16*)take(131072 * 2);
  __bf16* qk0h = (__bf16*)take((size_t)Bv * Hv * Nv * DHv * 2);
  __bf16* qk1h = (__bf16*)take((size_t)Bv * Hv * Nv * DHv * 2);
  __bf16* v0h  = (__bf16*)take((size_t)Bv * Hv * Nv * DHv * 2);
  __bf16* v1h  = (__bf16*)take((size_t)Bv * Hv * Nv * DHv * 2);
  __bf16* mm0  = (__bf16*)take((size_t)Bv * Nv * Dv * 2);
  __bf16* mm1  = (__bf16*)take((size_t)Bv * Nv * Dv * 2);
  __bf16* cat0 = (__bf16*)take((size_t)Bv * Nv * 512 * 2);
  __bf16* cat1 = (__bf16*)take((size_t)Bv * Nv * 512 * 2);
  float*  h1a  = (float*) take((size_t)Bv * Nv * 512 * 4);
  float*  h1b  = (float*) take((size_t)Bv * Nv * 512 * 4);
  __bf16* h2a  = (__bf16*)take((size_t)Bv * Nv * 512 * 2);
  __bf16* h2b  = (__bf16*)take((size_t)Bv * Nv * 512 * 2);
  float*  rmax = (float*) take((size_t)Bv * Hv * Nv * 4);
  float*  rinv = (float*) take((size_t)Bv * Hv * Nv * 4);
  float*  cmax = (float*) take((size_t)Bv * Hv * Nv * 4);
  float*  cinv = (float*) take((size_t)Bv * Hv * Nv * 4);
  if (off > ws_size) return;  // workspace too small — nothing safe to do

  const int M = Bv * Nv;  // 4096 rows

  // ---- stage bf16 copies (x0/x1 into LEFT half of concat buffers) ------
  cvt2bf<<<(M * Dv + 255) / 256, 256, 0, stream>>>(x0, cat0, Dv, 512, M * Dv);
  cvt2bf<<<(M * Dv + 255) / 256, 256, 0, stream>>>(x1, cat1, Dv, 512, M * Dv);
  cvt2bf<<<(65536 + 255) / 256, 256, 0, stream>>>(Wqk, wqkb, 256, 256, 65536);
  cvt2bf<<<(65536 + 255) / 256, 256, 0, stream>>>(Wv,  wvb,  256, 256, 65536);
  cvt2bf<<<(65536 + 255) / 256, 256, 0, stream>>>(Wo,  wob,  256, 256, 65536);
  cvt2bf<<<(262144 + 255) / 256, 256, 0, stream>>>(W1, w1b, 512, 512, 262144);
  cvt2bf<<<(131072 + 255) / 256, 256, 0, stream>>>(W2, w2b, 256, 256, 131072);

  // ---- projections: qk = (x@Wqk+bqk)*s, v = x@Wv+bv  (head-split bf16) --
  {
    int blocks = (M / 128) * (Dv >> 4);  // 512
    gemm_wmma<<<blocks, 256, 0, stream>>>(cat0, 512, wqkb, 256, bqk, nullptr, 0,
                                          qk0h, 0, 0, M, Dv, Dv, QK_SCALE, 2);
    gemm_wmma<<<blocks, 256, 0, stream>>>(cat1, 512, wqkb, 256, bqk, nullptr, 0,
                                          qk1h, 0, 0, M, Dv, Dv, QK_SCALE, 2);
    gemm_wmma<<<blocks, 256, 0, stream>>>(cat0, 512, wvb, 256, bv, nullptr, 0,
                                          v0h, 0, 0, M, Dv, Dv, 1.f, 2);
    gemm_wmma<<<blocks, 256, 0, stream>>>(cat1, 512, wvb, 256, bv, nullptr, 0,
                                          v1h, 0, 0, M, Dv, Dv, 1.f, 2);
  }

  // ---- sim = qk0 @ qk1^T  (raw scores into attn01 region) --------------
  qk_sim_wmma<<<(8 * 128 * 128) / 8, 256, 0, stream>>>(qk0h, qk1h, attn01);

  // ---- softmax statistics ----------------------------------------------
  row_softmax_stats<<<Bv * Hv * Nv, 256, 0, stream>>>(attn01, cm1, rmax, rinv);
  col_softmax_stats<<<Bv * Hv * (Nv / 64), 256, 0, stream>>>(attn01, cm0, cmax, cinv);

  // ---- finalize: attn01 in place, attn10 transposed --------------------
  softmax_finalize<<<Bv * Hv * 64 * 64, 256, 0, stream>>>(
      attn01, attn10, cm0, cm1, rmax, rinv, cmax, cinv);

  // ---- m0 = attn01 @ v1 ; m1 = attn10 @ v0  (merged heads -> mm) -------
  attn_v_wmma<<<512, 256, 0, stream>>>(attn01, v1h, mm0);
  attn_v_wmma<<<512, 256, 0, stream>>>(attn10, v0h, mm1);

  // ---- output projection: cat_right = mm @ Wo + bo ---------------------
  {
    int blocks = (M / 128) * (Dv >> 4);  // 512
    gemm_wmma<<<blocks, 256, 0, stream>>>(mm0, Dv, wob, 256, bo, nullptr, 0,
                                          cat0, 512, Dv, M, Dv, Dv, 1.f, 0);
    gemm_wmma<<<blocks, 256, 0, stream>>>(mm1, Dv, wob, 256, bo, nullptr, 0,
                                          cat1, 512, Dv, M, Dv, Dv, 1.f, 0);
  }

  // ---- FFN --------------------------------------------------------------
  {
    int blocks1 = (M / 128) * (512 >> 4);  // 1024
    gemm_wmma<<<blocks1, 256, 0, stream>>>(cat0, 512, w1b, 512, b1, nullptr, 0,
                                           h1a, 512, 0, M, 512, 512, 1.f, 1);
    gemm_wmma<<<blocks1, 256, 0, stream>>>(cat1, 512, w1b, 512, b1, nullptr, 0,
                                           h1b, 512, 0, M, 512, 512, 1.f, 1);
    ln_gelu<<<M, 256, 0, stream>>>(h1a, lng, lnb, h2a);
    ln_gelu<<<M, 256, 0, stream>>>(h1b, lng, lnb, h2b);
    int blocks2 = (M / 128) * (Dv >> 4);   // 512
    gemm_wmma<<<blocks2, 256, 0, stream>>>(h2a, 512, w2b, 256, b2, x0, Dv,
                                           out0, Dv, 0, M, Dv, 512, 1.f, 1);
    gemm_wmma<<<blocks2, 256, 0, stream>>>(h2b, 512, w2b, 256, b2, x1, Dv,
                                           out1, Dv, 0, M, Dv, 512, 1.f, 1);
  }
}
